// PlanEmbeddingNetPredVersion_13700945674658
// MI455X (gfx1250) — matrix operand order repaired and checked by reading, not compiled, not run on hardware
//
#include <hip/hip_runtime.h>
#include <hip/hip_bf16.h>
#include <cmath>
#include <cstdint>

// ---------------------------------------------------------------------------
// PlanEmbeddingNet forward for MI455X (gfx1250).
// bf16 WMMA tree-convs, one workgroup per batch, pipeline resident in LDS.
// K ordered tap-major (k = kk*Cpad + c), activations stored column-major
// X[j][c] so a B fragment is two ds_load_b128; A fragments are batched,
// ping-pong double-buffered, and forced onto the GLOBAL path (address_space(1))
// so A-waits (LOADcnt) decouple from B-waits (DScnt).
// ---------------------------------------------------------------------------

typedef __attribute__((ext_vector_type(16))) __bf16 v16bf;
typedef __attribute__((ext_vector_type(8)))  __bf16 v8bf;
typedef __attribute__((ext_vector_type(8)))  float  v8f;

#define GAS __attribute__((address_space(1)))

union AV { v16bf v; v8bf h[2]; };

// generic -> global(AS1) pointer (amdgcn: same address values for global mem)
template<typename T>
__device__ inline const T GAS* gptr(const T* p) {
  return (const T GAS*)(uintptr_t)p;
}
template<typename T>
__device__ inline T GAS* gptr_mut(T* p) {
  return (T GAS*)(uintptr_t)p;
}

#define MAXPRED 30

// Per-layer geometry: Cpad multiple of 32; Kpad = 3*Cpad; KT = Kpad/32
#define CPAD1 160   // real C_in 131
#define CPAD2 256
#define CPAD3 128
#define KPAD1 (3*CPAD1)   // 480, KT=15
#define KPAD2 (3*CPAD2)   // 768, KT=24
#define KPAD3 (3*CPAD3)   // 384, KT=12

// LDS arena (bytes)
#define OFF_IDX 0                        // int[128*3]
#define OFF_X1  2048                     // bf16[128][160] = 40960 (col-major)
#define OFF_X2  (OFF_X1 + 40960)         // bf16[128][256] = 65536
#define OFF_X3  (OFF_X2 + 65536)         // bf16[128][128] = 32768
#define OFF_Y3  (OFF_X3 + 32768)         // f32 [64][128]  = 32768 (row-major)
#define OFF_RED (OFF_Y3 + 32768)         // f32 scratch
#define SMEM_BYTES (OFF_RED + 512)       // 174,592 B (< 320 KB / WGP)

// ---------------------------------------------------------------------------
template<int CT>
__device__ inline void load_afrags(AV* av, const __bf16* __restrict__ Wg,
                                   const int Kpad, const int rowbase,
                                   const int r, const int lo, const int kt)
{
  #pragma unroll
  for (int ot = 0; ot < CT; ++ot) {
    const __bf16* p = Wg + (size_t)(rowbase + ot*16 + r) * Kpad + kt*32 + lo*16;
    const v8bf GAS* g = (const v8bf GAS*)(uintptr_t)p;   // force global_load_b128
    av[ot].h[0] = g[0];                     // K: kt*32 + lo*16 .. +7
    av[ot].h[1] = g[1];                     // K: kt*32 + lo*16 + 8 .. +15
  }
}

template<int Cpad>
__device__ inline v16bf load_bfrag(const __bf16* __restrict__ Xc,
                                   const int i0, const int i1, const int i2,
                                   const int kt, const int lo, const bool jz)
{
  AV t;
  if (jz) {                                  // reference's injected zero column
    #pragma unroll
    for (int e = 0; e < 16; ++e) t.v[e] = (__bf16)0.f;
  } else {
    const int kbase = kt*32 + lo*16;         // 16-elem span within one kk block
    const int kk  = kbase / Cpad;            // compile-time Cpad -> mul/shift
    const int c   = kbase - kk*Cpad;
    const int col = (kk == 0) ? i0 : ((kk == 1) ? i1 : i2);
    const v8bf* p = (const v8bf*)(Xc + (size_t)col*Cpad + c);
    t.h[0] = p[0];                           // two ds_load_b128
    t.h[1] = p[1];
  }
  return t.v;
}

// ---------------------------------------------------------------------------
// One GEMM pass over CT 16-row C_out tiles, full K, software-pipelined.
// Writes RAW conv+bias values to LDS and accumulates sum/sumsq into red[0..1].
// ---------------------------------------------------------------------------
template<int CT, int KT, int Cpad, bool OUTF32>
__device__ void gemm_pass(const __bf16* __restrict__ Wg, const int Kpad,
                          const int rowbase, const float* __restrict__ bias,
                          const __bf16* __restrict__ Xc,
                          const int* __restrict__ idxl,
                          __bf16* __restrict__ outBF, float* __restrict__ outF,
                          const int ldout, float* __restrict__ red, const int tid)
{
  const int lane = tid & 31;
  const int wave = tid >> 5;
  const int r    = lane & 15;       // A row-in-tile / B column-in-tile
  const int lo   = lane >> 4;       // ISA half select
  const int j    = wave*16 + r;     // output column 0..127
  const bool jz  = (j == 0);
  const int i0 = idxl[j*3], i1 = idxl[j*3+1], i2 = idxl[j*3+2];
  const float GAS* biasg = gptr(bias);

  v8f acc[CT];
  #pragma unroll
  for (int ot = 0; ot < CT; ++ot)
    acc[ot] = (v8f){0.f,0.f,0.f,0.f,0.f,0.f,0.f,0.f};

  AV a0[CT], a1[CT];
  v16bf b0, b1;
  load_afrags<CT>(a0, Wg, Kpad, rowbase, r, lo, 0);
  b0 = load_bfrag<Cpad>(Xc, i0, i1, i2, 0, lo, jz);
  b1 = b0;

  for (int kt = 0; kt < KT; kt += 2) {
    if (kt + 1 < KT) {                       // prefetch kt+1 while wmma(kt)
      load_afrags<CT>(a1, Wg, Kpad, rowbase, r, lo, kt + 1);
      b1 = load_bfrag<Cpad>(Xc, i0, i1, i2, kt + 1, lo, jz);
    }
    #pragma unroll
    for (int ot = 0; ot < CT; ++ot)
      acc[ot] = __builtin_amdgcn_wmma_f32_16x16x32_bf16(
          false, a0[ot].v, false, b0, (short)0, acc[ot], false, false);
    if (kt + 2 < KT) {                       // prefetch kt+2 while wmma(kt+1)
      load_afrags<CT>(a0, Wg, Kpad, rowbase, r, lo, kt + 2);
      b0 = load_bfrag<Cpad>(Xc, i0, i1, i2, kt + 2, lo, jz);
    }
    if (kt + 1 < KT) {
      #pragma unroll
      for (int ot = 0; ot < CT; ++ot)
        acc[ot] = __builtin_amdgcn_wmma_f32_16x16x32_bf16(
            false, a1[ot].v, false, b1, (short)0, acc[ot], false, false);
    }
  }

  // ---- raw store (pre-norm) + stats ----
  float s = 0.f, s2 = 0.f;
  #pragma unroll
  for (int ot = 0; ot < CT; ++ot) {
    #pragma unroll
    for (int e = 0; e < 8; ++e) {
      const int row = rowbase + ot*16 + lo*8 + e;   // C/D layout: M = lo*8+e
      float v = acc[ot][e];
      v = jz ? 0.f : (v + biasg[row]);
      if (OUTF32) outF[row*128 + j] = v;                    // row-major f32
      else        outBF[(size_t)j*ldout + row] = (__bf16)v; // col-major bf16
      s += v; s2 += v*v;
    }
  }
  #pragma unroll
  for (int off = 16; off > 0; off >>= 1) {
    s  += __shfl_xor(s,  off, 32);
    s2 += __shfl_xor(s2, off, 32);
  }
  if (lane == 0) { atomicAdd(&red[0], s); atomicAdd(&red[1], s2); }
}

// ---------------------------------------------------------------------------
__global__ void __launch_bounds__(256)
plan_embed_kernel(const float* __restrict__ feature,    // [B,128,128]
                  const int*   __restrict__ indexes,    // [B,381]
                  const float* __restrict__ col_embed,  // [200,32]
                  const float* __restrict__ op_embed,   // [20,32]
                  const float* __restrict__ b1,
                  const float* __restrict__ b2,
                  const float* __restrict__ b3,
                  const float* __restrict__ fw1, const float* __restrict__ fb1,
                  const float* __restrict__ fw2, const float* __restrict__ fb2,
                  const __bf16* __restrict__ W1,
                  const __bf16* __restrict__ W2,
                  const __bf16* __restrict__ W3,
                  float* __restrict__ out)
{
  extern __shared__ char smem[];
  int*    idxl = (int*)   (smem + OFF_IDX);
  __bf16* X1c  = (__bf16*)(smem + OFF_X1);   // [j][160]
  __bf16* X2c  = (__bf16*)(smem + OFF_X2);   // [j][256]
  __bf16* X3c  = (__bf16*)(smem + OFF_X3);   // [j][128]
  float*  Y3   = (float*) (smem + OFF_Y3);   // [row][128]
  float*  red  = (float*) (smem + OFF_RED);

  const int b   = blockIdx.x;
  const int tid = threadIdx.x;
  const float GAS* F = gptr(feature) + (size_t)b * 128 * 128;

  // ---- indexes -> LDS (j=0 is the zero column: dummy indices) ----
  if (tid < 128) {
    const int GAS* idxg = gptr(indexes);
    if (tid == 0) { idxl[0] = 0; idxl[1] = 0; idxl[2] = 0; }
    else {
      const int base = b * 381 + 3*(tid - 1);
      idxl[tid*3+0] = idxg[base+0];
      idxl[tid*3+1] = idxg[base+1];
      idxl[tid*3+2] = idxg[base+2];
    }
  }

  // ---- collate: others | masked embed sums | param -> X1c (col-major bf16) --
  {
    const int i = tid & 127;           // node
    const int h = tid >> 7;            // 0: col_embed half, 1: op_embed half
    __bf16* xrow = X1c + (size_t)i * CPAD1;
    float accv[32];
    #pragma unroll
    for (int e = 0; e < 32; ++e) accv[e] = 0.f;
    const float len = F[95*128 + i];   // length row = OTHER + 2*MAX_PRED
    const float GAS* etab = h ? gptr(op_embed) : gptr(col_embed);
    for (int p = 0; p < MAXPRED; ++p) {
      if ((float)p < len) {
        const int id = (int)F[(35 + h*30 + p)*128 + i];
        const float GAS* ev = etab + id*32;
        #pragma unroll
        for (int e = 0; e < 32; ++e) accv[e] += ev[e];
      }
    }
    #pragma unroll
    for (int e = 0; e < 32; ++e) xrow[35 + h*32 + e] = (__bf16)accv[e];
    if (h == 0) {
      for (int c = 0; c < 35; ++c) xrow[c] = (__bf16)F[c*128 + i];       // others
    } else {
      #pragma unroll
      for (int e = 0; e < 32; ++e) xrow[99 + e] = (__bf16)F[(96+e)*128 + i]; // param
    }
    for (int c = 131 + h; c < CPAD1; c += 2) xrow[c] = (__bf16)0.f;      // K-pad
  }

  // ================= layer 1: 131->256, leaky =================
  __syncthreads();
  if (tid < 2) red[tid] = 0.f;
  __syncthreads();
  gemm_pass<8, KPAD1/32, CPAD1, false>(W1, KPAD1, 0,   b1, X1c, idxl, X2c, nullptr, CPAD2, red, tid);
  gemm_pass<8, KPAD1/32, CPAD1, false>(W1, KPAD1, 128, b1, X1c, idxl, X2c, nullptr, CPAD2, red, tid);
  __syncthreads();
  {
    const float n = 256.f * 128.f;
    const float mean = red[0] / n;
    const float var  = (red[1] - n*mean*mean) / (n - 1.f);
    const float inv  = 1.f / (sqrtf(var) + 1e-5f);
    for (int i = tid; i < 256*128; i += 256) {
      float v = ((float)X2c[i] - mean) * inv;
      X2c[i] = (__bf16)(v > 0.f ? v : 0.01f*v);
    }
  }

  // ================= layer 2: 256->128, leaky =================
  __syncthreads();
  if (tid < 2) red[tid] = 0.f;
  __syncthreads();
  gemm_pass<8, KPAD2/32, CPAD2, false>(W2, KPAD2, 0, b2, X2c, idxl, X3c, nullptr, CPAD3, red, tid);
  __syncthreads();
  {
    const float n = 128.f * 128.f;
    const float mean = red[0] / n;
    const float var  = (red[1] - n*mean*mean) / (n - 1.f);
    const float inv  = 1.f / (sqrtf(var) + 1e-5f);
    for (int i = tid; i < 128*128; i += 256) {
      float v = ((float)X3c[i] - mean) * inv;
      X3c[i] = (__bf16)(v > 0.f ? v : 0.01f*v);
    }
  }

  // ================= layer 3: 128->64, norm only ==============
  __syncthreads();
  if (tid < 2) red[tid] = 0.f;
  __syncthreads();
  gemm_pass<4, KPAD3/32, CPAD3, true>(W3, KPAD3, 0, b3, X3c, idxl, nullptr, Y3, 0, red, tid);
  __syncthreads();
  {
    const float n = 64.f * 128.f;
    const float mean = red[0] / n;
    const float var  = (red[1] - n*mean*mean) / (n - 1.f);
    const float inv  = 1.f / (sqrtf(var) + 1e-5f);
    for (int i = tid; i < 64*128; i += 256) Y3[i] = (Y3[i] - mean) * inv;
  }
  __syncthreads();

  // ---- max pool over nodes (zero column included, as in reference) ----
  if (tid < 64) {
    float m = -INFINITY;
    for (int jj = 0; jj < 128; ++jj) m = fmaxf(m, Y3[tid*128 + jj]);
    red[8 + tid] = m;
  }
  __syncthreads();
  // ---- tiny MLP head in fp32 ----
  if (tid < 32) {
    float hv = gptr(fb1)[tid];
    const float GAS* w1g = gptr(fw1);
    for (int c = 0; c < 64; ++c) hv += red[8 + c] * w1g[tid*64 + c];
    red[72 + tid] = fmaxf(hv, 0.f);
  }
  __syncthreads();
  if (tid == 0) {
    float o = gptr(fb2)[0];
    const float GAS* w2g = gptr(fw2);
    for (int i2 = 0; i2 < 32; ++i2) o += red[72 + i2] * w2g[i2];
    gptr_mut(out)[b] = o;
  }
}

// ---------------------------------------------------------------------------
// Pack fp32 conv weights w[o][c][kk] -> bf16 Wp[o][kk*Cpad + c], zero-padded.
// ---------------------------------------------------------------------------
__global__ void pack_weights(const float* __restrict__ w, __bf16* __restrict__ dst,
                             int total, int Creal, int Cpad)
{
  const int i = blockIdx.x * blockDim.x + threadIdx.x;
  if (i >= total) return;
  const int K   = 3 * Cpad;
  const int o   = i / K;
  const int rem = i - o * K;
  const int kk  = rem / Cpad;
  const int c   = rem - kk * Cpad;
  gptr_mut(dst)[i] = (c < Creal)
      ? (__bf16)gptr(w)[((size_t)o * Creal + c) * 3 + kk] : (__bf16)0.f;
}

// ---------------------------------------------------------------------------
extern "C" void kernel_launch(void* const* d_in, const int* in_sizes, int n_in,
                              void* d_out, int out_size, void* d_ws, size_t ws_size,
                              hipStream_t stream)
{
  const float* feature   = (const float*)d_in[0];
  const int*   indexes   = (const int*)  d_in[1];   // jax default x64-off => int32
  const float* col_embed = (const float*)d_in[2];
  const float* op_embed  = (const float*)d_in[3];
  const float* w1  = (const float*)d_in[4];
  const float* b1  = (const float*)d_in[5];
  const float* w2  = (const float*)d_in[6];
  const float* b2  = (const float*)d_in[7];
  const float* w3  = (const float*)d_in[8];
  const float* b3  = (const float*)d_in[9];
  const float* fw1 = (const float*)d_in[10];
  const float* fb1 = (const float*)d_in[11];
  const float* fw2 = (const float*)d_in[12];
  const float* fb2 = (const float*)d_in[13];

  // workspace: packed bf16 weights (stay hot in the 192MB L2 across blocks)
  const size_t s1 = (size_t)256 * KPAD1 * 2;   // 245,760
  const size_t s2 = (size_t)128 * KPAD2 * 2;   // 196,608
  __bf16* W1 = (__bf16*)d_ws;
  __bf16* W2 = (__bf16*)((char*)d_ws + s1);
  __bf16* W3 = (__bf16*)((char*)d_ws + s1 + s2);

  const int t1 = 256 * KPAD1, t2 = 128 * KPAD2, t3 = 64 * KPAD3;
  pack_weights<<<(t1 + 255) / 256, 256, 0, stream>>>(w1, W1, t1, 131, CPAD1);
  pack_weights<<<(t2 + 255) / 256, 256, 0, stream>>>(w2, W2, t2, 256, CPAD2);
  pack_weights<<<(t3 + 255) / 256, 256, 0, stream>>>(w3, W3, t3, 128, CPAD3);

  plan_embed_kernel<<<256, 256, SMEM_BYTES, stream>>>(
      feature, indexes, col_embed, op_embed, b1, b2, b3,
      fw1, fb1, fw2, fb2, W1, W2, W3, (float*)d_out);
}
